// Attention_28329604285133
// MI455X (gfx1250) — matrix operand order repaired and checked by reading, not compiled, not run on hardware
//
#include <hip/hip_runtime.h>
#include <hip/hip_bf16.h>

// ---------------------------------------------------------------------------
// Problem constants (from reference): B=4, S=2048, HID=2048, NH=16, HD=128
// All GEMMs are 8192 x 2048 x 2048 -> hardcode so row offsets fold into the
// 24-bit instruction offset field (cuts v_add_nc_u64 chains in epilogues).
// ---------------------------------------------------------------------------
#define BB   4
#define SS   2048
#define HID  2048
#define NH   16
#define HD   128
#define MM   (BB * SS)          // 8192 GEMM rows

typedef __bf16 bf16;
typedef bf16  v16bf __attribute__((ext_vector_type(16)));
typedef bf16  v8bf  __attribute__((ext_vector_type(8)));
typedef bf16  v4bf  __attribute__((ext_vector_type(4)));
typedef float v8f   __attribute__((ext_vector_type(8)));

// Two 16-byte LDS loads -> one 16x bf16 WMMA operand (8 VGPRs/lane).
static __device__ __forceinline__ v16bf load_frag(const bf16* p0, const bf16* p1) {
    union { v16bf v; v8bf h[2]; } u;
    u.h[0] = *(const v8bf*)p0;
    u.h[1] = *(const v8bf*)p1;
    return u.v;
}

static __device__ __forceinline__ v4bf cvt4(float4 f) {
    v4bf r;
    r[0] = (bf16)f.x; r[1] = (bf16)f.y; r[2] = (bf16)f.z; r[3] = (bf16)f.w;
    return r;
}

static __device__ __forceinline__ v8f wmma_bf16(v16bf a, v16bf b, v8f c) {
    // (neg_a, A, neg_b, B, c_mod, C, reuse_a, reuse_b)
    return __builtin_amdgcn_wmma_f32_16x16x32_bf16(false, a, false, b, (short)0, c,
                                                   false, false);
}

static __device__ __forceinline__ float hmax16(float v) {
    for (int m = 1; m < 16; m <<= 1) v = fmaxf(v, __shfl_xor(v, m, 16));
    return v;
}
static __device__ __forceinline__ float hsum16(float v) {
    for (int m = 1; m < 16; m <<= 1) v += __shfl_xor(v, m, 16);
    return v;
}

// ---------------------------------------------------------------------------
// f32 -> bf16-WMMA GEMM:  Y[8192,2048] = X[8192,2048] * W[2048,2048]
// 256 threads (8 waves). Block tile 128x128, wave tile 32x64, K step 64
// (two WMMA K-chunks per stage -> 16 WMMAs per barrier interval).
// LDS: X tile [128][64] row-major, W tile transposed [128][72] (pad keeps
// 16B fragment alignment and staggers store banks).
// ---------------------------------------------------------------------------
#define WT_STRIDE 72

__global__ __launch_bounds__(256) void gemm_bf16_wmma(
    const float* __restrict__ X, const float* __restrict__ W,
    float* __restrict__ Y) {
    __shared__ bf16 xs[128 * 64];          // 16 KB
    __shared__ bf16 wt[128 * WT_STRIDE];   // 18 KB

    const int tid  = threadIdx.x;
    const int lane = tid & 31;
    const int wave = tid >> 5;
    const int hi   = lane >> 4;
    const int ln   = lane & 15;

    const int n0 = blockIdx.x * 128;
    const int m0 = blockIdx.y * 128;
    const int wm = (wave & 3) * 32;
    const int wn = (wave >> 2) * 64;

    v8f acc[2][4] = {};

    for (int k0 = 0; k0 < HID; k0 += 64) {
        // prefetch next K tile into GL2 (gfx1250 global_prefetch_b8)
        if (k0 + 64 < HID) {
            __builtin_prefetch(&X[(size_t)(m0 + (tid >> 1)) * HID + k0 + 64], 0, 1);
            __builtin_prefetch(&W[(size_t)(k0 + 64 + (tid >> 4)) * HID + n0 + (tid & 15) * 8], 0, 1);
        }
        // stage X tile 128x64 as float4 (global_load_b128), ds_store_b64
        #pragma unroll
        for (int i = 0; i < 8; ++i) {
            int idx = tid + i * 256;            // 0..2047 float4s
            int r = idx >> 4, c4 = idx & 15;    // 16 float4 per row
            float4 f = *(const float4*)&X[(size_t)(m0 + r) * HID + k0 + c4 * 4];
            *(v4bf*)&xs[r * 64 + c4 * 4] = cvt4(f);
        }
        // stage W tile 64x128 as float4, transpose into [n][k] with pad
        #pragma unroll
        for (int i = 0; i < 8; ++i) {
            int idx = tid + i * 256;
            int kr = idx >> 5, nc4 = idx & 31;  // 32 float4 per row
            float4 f = *(const float4*)&W[(size_t)(k0 + kr) * HID + n0 + nc4 * 4];
            bf16* wp = &wt[(nc4 * 4) * WT_STRIDE + kr];
            wp[0]             = (bf16)f.x;
            wp[WT_STRIDE]     = (bf16)f.y;
            wp[2 * WT_STRIDE] = (bf16)f.z;
            wp[3 * WT_STRIDE] = (bf16)f.w;
        }
        __syncthreads();

        #pragma unroll
        for (int kc = 0; kc < 2; ++kc) {
            v16bf a[2], b[4];
            #pragma unroll
            for (int mi = 0; mi < 2; ++mi) {
                const bf16* p = xs + (wm + mi * 16 + ln) * 64 + kc * 32 + hi * 16;
                a[mi] = load_frag(p, p + 8);    // K {0..15}/{16..31}
            }
            #pragma unroll
            for (int ni = 0; ni < 4; ++ni) {
                const bf16* p = wt + (wn + ni * 16 + ln) * WT_STRIDE + kc * 32 + hi * 16;
                b[ni] = load_frag(p, p + 8);    // K {0..15}/{16..31}
            }
            #pragma unroll
            for (int mi = 0; mi < 2; ++mi)
                #pragma unroll
                for (int ni = 0; ni < 4; ++ni)
                    acc[mi][ni] = wmma_bf16(a[mi], b[ni], acc[mi][ni]);
        }
        __syncthreads();
    }

    // C layout: VGPR r -> row r (lanes 0-15) / row 8+r (lanes 16-31), col = ln
    #pragma unroll
    for (int mi = 0; mi < 2; ++mi)
        #pragma unroll
        for (int ni = 0; ni < 4; ++ni) {
            float* yb = Y + (size_t)(m0 + wm + mi * 16 + hi * 8) * HID +
                        n0 + wn + ni * 16 + ln;
            #pragma unroll
            for (int r = 0; r < 8; ++r)
                yb[r * HID] = acc[mi][ni][r];   // imm-offset stores
        }
}

// ---------------------------------------------------------------------------
// Fused RMSNorm (per-head, HD=128) + RoPE, in place on Q or K buffer.
// One wave per (b,s,h) vector; lane owns {l, l+32, l+64, l+96} so the
// rotate-half pairs (i, i+64) stay lane-local.
// ---------------------------------------------------------------------------
__global__ __launch_bounds__(256) void rmsnorm_rope_kernel(
    float* __restrict__ x, const float* __restrict__ w,
    const float* __restrict__ cosT, const float* __restrict__ sinT) {
    const int lane = threadIdx.x & 31;
    const int wave = threadIdx.x >> 5;
    const int vid  = blockIdx.x * 8 + wave;        // (b*S+s)*NH + h
    const int s    = (vid >> 4) & (SS - 1);

    float* p = x + (size_t)vid * HD;
    float x0 = p[lane], x1 = p[lane + 32], x2 = p[lane + 64], x3 = p[lane + 96];

    float ssq = x0 * x0 + x1 * x1 + x2 * x2 + x3 * x3;
    for (int m = 1; m < 32; m <<= 1) ssq += __shfl_xor(ssq, m, 32);
    float rn = rsqrtf(ssq * (1.0f / HD) + 1e-6f);

    float y0 = x0 * rn * w[lane];
    float y1 = x1 * rn * w[lane + 32];
    float y2 = x2 * rn * w[lane + 64];
    float y3 = x3 * rn * w[lane + 96];

    const float* c  = cosT + (size_t)s * HD;
    const float* sn = sinT + (size_t)s * HD;
    p[lane]      = y0 * c[lane]      - y2 * sn[lane];
    p[lane + 32] = y1 * c[lane + 32] - y3 * sn[lane + 32];
    p[lane + 64] = y2 * c[lane + 64] + y0 * sn[lane + 64];
    p[lane + 96] = y3 * c[lane + 96] + y1 * sn[lane + 96];
}

// ---------------------------------------------------------------------------
// Flash-style attention. One block per (b, h, 128-row Q tile); 8 waves,
// each wave owns 16 Q rows (Q fragments resident in VGPRs) and streams
// 32-key tiles: QK^T (8 WMMA) -> online softmax (half-wave shuffles) ->
// P re-layout through per-wave LDS -> P*V (8 WMMA).
// ---------------------------------------------------------------------------
__global__ __launch_bounds__(256) void flash_attn_kernel(
    const float* __restrict__ Q, const float* __restrict__ Kp,
    const float* __restrict__ Vp, const float* __restrict__ mask,
    float* __restrict__ Out) {
    __shared__ bf16 qs[128 * 128];   // 32 KB  Q tile  [row][hd]
    __shared__ bf16 ks[32 * 128];    //  8 KB  K tile  [key][hd]
    __shared__ bf16 vts[128 * 32];   //  8 KB  V tile  [hd][key] (transposed)
    __shared__ bf16 ps[8][16 * 32];  //  8 KB  per-wave P scratch [row][key]

    const int tid  = threadIdx.x;
    const int lane = tid & 31;
    const int wave = tid >> 5;
    const int hi   = lane >> 4;
    const int ln   = lane & 15;

    const int qb = blockIdx.x & 15;          // 16 q-tiles of 128
    const int h  = (blockIdx.x >> 4) & 15;
    const int b  = blockIdx.x >> 8;

    const size_t base = ((size_t)b * SS) * HID + (size_t)h * HD; // + row*HID

    // stage Q tile (float4 loads, ds_store_b64)
    #pragma unroll
    for (int i = 0; i < 16; ++i) {
        int idx = tid + i * 256;             // 0..4095 float4s
        int r = idx >> 5, c4 = idx & 31;     // 32 float4 per row
        float4 f = *(const float4*)&Q[base + (size_t)(qb * 128 + r) * HID + c4 * 4];
        *(v4bf*)&qs[r * 128 + c4 * 4] = cvt4(f);
    }
    __syncthreads();

    v16bf aq[4];
    #pragma unroll
    for (int kc = 0; kc < 4; ++kc) {
        const bf16* p = qs + (wave * 16 + ln) * 128 + kc * 32 + hi * 16;
        aq[kc] = load_frag(p, p + 8);
    }

    v8f o[8] = {};
    float rm[8], rl[8];
    #pragma unroll
    for (int r = 0; r < 8; ++r) { rm[r] = -__builtin_inff(); rl[r] = 0.0f; }
    const float scale = 0.08838834764831845f;   // HD^-0.5

    for (int kt = 0; kt < SS; kt += 32) {
        __syncthreads();                         // previous readers done
        #pragma unroll
        for (int i = 0; i < 4; ++i) {
            int idx = tid + i * 256;             // 0..1023 float4s
            int r = idx >> 5, c4 = idx & 31;
            size_t g = base + (size_t)(kt + r) * HID + c4 * 4;
            float4 fk = *(const float4*)&Kp[g];
            float4 fv = *(const float4*)&Vp[g];
            *(v4bf*)&ks[r * 128 + c4 * 4] = cvt4(fk);
            bf16* vp = &vts[(c4 * 4) * 32 + r];
            vp[0]  = (bf16)fv.x;
            vp[32] = (bf16)fv.y;
            vp[64] = (bf16)fv.z;
            vp[96] = (bf16)fv.w;
        }
        __syncthreads();

        // scores: 16 q-rows x 32 keys, K-dim = HD in 4 chunks of 32
        v8f s0 = {}, s1 = {};
        #pragma unroll
        for (int kc = 0; kc < 4; ++kc) {
            const bf16* p0 = ks + ln * 128 + kc * 32 + hi * 16;
            const bf16* p1 = ks + (16 + ln) * 128 + kc * 32 + hi * 16;
            v16bf bk0 = load_frag(p0, p0 + 8);
            v16bf bk1 = load_frag(p1, p1 + 8);
            s0 = wmma_bf16(aq[kc], bk0, s0);
            s1 = wmma_bf16(aq[kc], bk1, s1);
        }

        // scale + attention mask (B,1,S,S), imm-offset row addressing
        const float* mp =
            mask + ((size_t)b * SS + (qb * 128 + wave * 16 + hi * 8)) * SS + kt + ln;
        #pragma unroll
        for (int r = 0; r < 8; ++r) {
            s0[r] = s0[r] * scale + mp[r * SS];
            s1[r] = s1[r] * scale + mp[r * SS + 16];
        }

        // online softmax; each half-wave holds one score row per VGPR index
        bf16* pw = ps[wave];
        #pragma unroll
        for (int r = 0; r < 8; ++r) {
            float mx   = hmax16(fmaxf(s0[r], s1[r]));
            float mnew = fmaxf(rm[r], mx);
            float alph = __expf(rm[r] - mnew);
            rm[r] = mnew;
            float p0 = __expf(s0[r] - mnew);
            float p1 = __expf(s1[r] - mnew);
            rl[r] = rl[r] * alph + hsum16(p0 + p1);
            #pragma unroll
            for (int nt = 0; nt < 8; ++nt) o[nt][r] *= alph;
            pw[(hi * 8 + r) * 32 + ln]      = (bf16)p0;
            pw[(hi * 8 + r) * 32 + 16 + ln] = (bf16)p1;
        }
        asm volatile("s_wait_dscnt 0" ::: "memory");  // P stores visible in-wave

        // O += P(16x32) * V(32x128)
        const bf16* app = pw + ln * 32 + hi * 16;
        v16bf ap = load_frag(app, app + 8);
        #pragma unroll
        for (int nt = 0; nt < 8; ++nt) {
            const bf16* bp = vts + (nt * 16 + ln) * 32 + hi * 16;
            v16bf bv = load_frag(bp, bp + 8);
            o[nt] = wmma_bf16(ap, bv, o[nt]);
        }
    }

    // normalize and write [b][s][h*HD + hd] (imm-offset stores)
    float* ob = Out + base + (size_t)(qb * 128 + wave * 16 + hi * 8) * HID + ln;
    #pragma unroll
    for (int r = 0; r < 8; ++r) {
        float inv = 1.0f / rl[r];
        #pragma unroll
        for (int nt = 0; nt < 8; ++nt)
            ob[r * HID + nt * 16] = o[nt][r] * inv;
    }
}

// ---------------------------------------------------------------------------
// Host-side launch
// ---------------------------------------------------------------------------
extern "C" void kernel_launch(void* const* d_in, const int* in_sizes, int n_in,
                              void* d_out, int out_size, void* d_ws, size_t ws_size,
                              hipStream_t stream) {
    const float* hs   = (const float*)d_in[0];
    const float* cosT = (const float*)d_in[1];
    const float* sinT = (const float*)d_in[2];
    const float* mask = (const float*)d_in[3];
    const float* wq   = (const float*)d_in[4];
    const float* wk   = (const float*)d_in[5];
    const float* wv   = (const float*)d_in[6];
    const float* wo   = (const float*)d_in[7];
    const float* qnw  = (const float*)d_in[8];
    const float* knw  = (const float*)d_in[9];

    const size_t chunk = (size_t)MM * HID;   // elements per scratch buffer
    float* q    = (float*)d_ws;
    float* k    = q + chunk;
    float* v    = k + chunk;
    float* attn = v + chunk;

    dim3 gg(HID / 128, MM / 128);            // (16, 64)

    // QKV projections
    gemm_bf16_wmma<<<gg, 256, 0, stream>>>(hs, wq, q);
    gemm_bf16_wmma<<<gg, 256, 0, stream>>>(hs, wk, k);
    gemm_bf16_wmma<<<gg, 256, 0, stream>>>(hs, wv, v);

    // per-head RMSNorm + RoPE (in place), 8 head-vectors per block
    int nvec = BB * SS * NH;                 // 131072
    rmsnorm_rope_kernel<<<nvec / 8, 256, 0, stream>>>(q, qnw, cosT, sinT);
    rmsnorm_rope_kernel<<<nvec / 8, 256, 0, stream>>>(k, knw, cosT, sinT);

    // attention: B * NH * (S/128) blocks
    flash_attn_kernel<<<BB * NH * (SS / 128), 256, 0, stream>>>(q, k, v, mask, attn);

    // output projection
    gemm_bf16_wmma<<<gg, 256, 0, stream>>>(attn, wo, (float*)d_out);
}
